// Model_51187420233804
// MI455X (gfx1250) — matrix operand order repaired and checked by reading, not compiled, not run on hardware
//
#include <hip/hip_runtime.h>
#include <hip/hip_bf16.h>
#include <math.h>

// ---------- types ----------
typedef __bf16 bf16_t;
typedef bf16_t bf16x8  __attribute__((ext_vector_type(8)));
typedef bf16_t bf16x16 __attribute__((ext_vector_type(16)));
typedef float  f32x8   __attribute__((ext_vector_type(8)));
typedef unsigned int v4u __attribute__((ext_vector_type(4)));
typedef int          v8i __attribute__((ext_vector_type(8)));
typedef int          v4i __attribute__((ext_vector_type(4)));

// Problem constants (from the reference)
#define NB   2048          // batch
#define NL   64            // words per sentence
#define ND   300           // embedding dim == conv kernel width (K of GEMM 1)
#define NC   512           // conv out channels
#define NH   2048          // fc hidden
#define MROWS (NB * NL)    // 131072 rows of GEMM 1

#define TM   128           // workgroup M tile (== 2 sentences)
#define TN   128           // workgroup N tile
#define TK   32            // WMMA K per step (bf16)
#define KPAD 40            // LDS row stride in bf16 (80B: 16B-aligned, conflict-skewed)

__device__ __forceinline__ f32x8 wmma_bf16(bf16x16 a, bf16x16 b, f32x8 c) {
    // D = A(16x32 bf16) x B(32x16 bf16) + C(16x16 f32)
    return __builtin_amdgcn_wmma_f32_16x16x32_bf16(
        /*neg_a=*/false, a, /*neg_b=*/false, b,
        /*c_mod=*/(short)0, c, /*reuse_a=*/false, /*reuse_b=*/false);
}

__device__ __forceinline__ bf16x16 cat8(bf16x8 lo, bf16x8 hi) {
    return __builtin_shufflevector(lo, hi, 0,1,2,3,4,5,6,7,8,9,10,11,12,13,14,15);
}

// =====================================================================
// TDM: 2D tile load Global -> LDS (bf16, tile = rows x 32, row-padded
// in LDS by 16B so LDS row stride == KPAD*2 == 80B).
// D# layout per CDNA5 ISA ch.8 (group0: control/lds/global addr,
// group1: dims/strides/pad, groups 2-3 unused for 2D -> zero).
// This toolchain exposes the 6-arg builtin form.
// =====================================================================
__device__ __forceinline__ void tdm_load_tile32(uint32_t lds_addr, const bf16_t* gptr,
                                                uint32_t k_len, uint32_t n_rows) {
    union { v4u v; uint32_t w[4]; } g0;
    union { v8i v; uint32_t w[8]; } g1;
    const uint64_t ga = (uint64_t)(uintptr_t)gptr;

    g0.w[0] = 1u;                                  // count=1, user descriptor
    g0.w[1] = lds_addr;                            // LDS byte offset of tile
    g0.w[2] = (uint32_t)ga;                        // global_addr[31:0]
    g0.w[3] = (uint32_t)((ga >> 32) & 0x01FFFFFFu) // global_addr[56:32]
            | 0x80000000u;                         // type=2 ("image")

    g1.w[0] = (1u << 16)                           // data_size = 2 bytes
            | (1u << 20)                           // pad_enable
            | (3u << 22)                           // pad_interval: 16 DWORDs (=64B = one 32-elem row)
            | (3u << 25);                          // pad_amount: 4 DWORDs (=16B)
    g1.w[1] = (k_len & 0xFFFFu) << 16;             // tensor_dim0[15:0]
    g1.w[2] = (k_len >> 16) | ((n_rows & 0xFFFFu) << 16);   // dim0 hi | tensor_dim1 lo
    g1.w[3] = (n_rows >> 16) | ((uint32_t)TK << 16);        // dim1 hi | tile_dim0 = 32
    g1.w[4] = (uint32_t)TM;                        // tile_dim1 = 128, tile_dim2 = 0
    g1.w[5] = k_len;                               // tensor_dim0_stride[31:0]
    g1.w[6] = 0u;                                  // stride hi | dim1_stride lo
    g1.w[7] = 0u;

    v4i z4{};
    v8i z8{};
    __builtin_amdgcn_tensor_load_to_lds(g0.v, g1.v, z4, z4, z8, /*cpol=*/0);
}

__device__ __forceinline__ uint32_t lds_off(const void* p) {
    // generic pointer to LDS: low 32 bits of the flat address are the LDS offset
    return (uint32_t)(uintptr_t)p;
}

// =====================================================================
// Stage 1: per-word conv (full-width filter == GEMM [M,300]x[300,512]),
// + bias + ReLU + max over the 64 words of each sentence, fused.
// grid = (NC/TN, MROWS/TM, 2)  -- N fastest so x-tile re-reads hit L2
// block = 256 (8 waves)
// =====================================================================
__global__ __launch_bounds__(256) void conv_relu_max_kernel(
    const float* __restrict__ x1, const float* __restrict__ x2,
    const float* __restrict__ Wc, const float* __restrict__ bc,
    float* __restrict__ e1, float* __restrict__ e2)
{
    __shared__ bf16_t As[TM][KPAD];
    __shared__ bf16_t Bs[TN][KPAD];
    __shared__ float  red[8][TN];

    const float* __restrict__ x = (blockIdx.z == 0) ? x1 : x2;
    float*       __restrict__ e = (blockIdx.z == 0) ? e1 : e2;

    const int tid  = threadIdx.x;
    const int w    = tid >> 5;          // wave id 0..7
    const int l    = tid & 31;          // lane
    const int lr   = l & 15;
    const int half = l >> 4;            // 0: lanes 0-15, 1: lanes 16-31
    const int n_base = blockIdx.x * TN; // channel block in [0, NC)   (fast axis)
    const int m_base = blockIdx.y * TM; // row block in [0, MROWS)

    f32x8 acc[8];
    #pragma unroll
    for (int nf = 0; nf < 8; ++nf) acc[nf] = f32x8{};

    const int KSTEPS = (ND + TK - 1) / TK; // 10 (300 padded to 320)
    for (int kt = 0; kt < KSTEPS; ++kt) {
        const int k0 = kt * TK;

        // ---- stage A tile [TM x 32] f32 -> bf16 LDS (zero-pad d >= 300) ----
        for (int idx = tid; idx < TM * 8; idx += 256) {
            const int r = idx >> 3;
            const int q = (idx & 7) * 4;
            const int d = k0 + q;
            const float* src = x + (size_t)(m_base + r) * ND + d;
            float4 v;
            if (d + 3 < ND)      v = *(const float4*)src;
            else {
                v.x = (d + 0 < ND) ? src[0] : 0.f;
                v.y = (d + 1 < ND) ? src[1] : 0.f;
                v.z = (d + 2 < ND) ? src[2] : 0.f;
                v.w = (d + 3 < ND) ? src[3] : 0.f;
            }
            As[r][q + 0] = (bf16_t)v.x; As[r][q + 1] = (bf16_t)v.y;
            As[r][q + 2] = (bf16_t)v.z; As[r][q + 3] = (bf16_t)v.w;
        }
        // ---- stage B tile: Wc rows (channels) [TN x 32] f32 -> bf16 LDS ----
        for (int idx = tid; idx < TN * 8; idx += 256) {
            const int r = idx >> 3;
            const int q = (idx & 7) * 4;
            const int d = k0 + q;
            const float* src = Wc + (size_t)(n_base + r) * ND + d;
            float4 v;
            if (d + 3 < ND)      v = *(const float4*)src;
            else {
                v.x = (d + 0 < ND) ? src[0] : 0.f;
                v.y = (d + 1 < ND) ? src[1] : 0.f;
                v.z = (d + 2 < ND) ? src[2] : 0.f;
                v.w = (d + 3 < ND) ? src[3] : 0.f;
            }
            Bs[r][q + 0] = (bf16_t)v.x; Bs[r][q + 1] = (bf16_t)v.y;
            Bs[r][q + 2] = (bf16_t)v.z; Bs[r][q + 3] = (bf16_t)v.w;
        }
        __syncthreads();

        // ---- A fragment: lane half 0 -> K 0-7 & 16-23, half 1 -> K 8-15 & 24-31
        const int akb = half ? 8 : 0;
        bf16x8 alo = *(const bf16x8*)&As[w * 16 + lr][akb];
        bf16x8 ahi = *(const bf16x8*)&As[w * 16 + lr][akb + 16];
        bf16x16 afrag = cat8(alo, ahi);

        // ---- B fragments: lane half 0 -> K 0-7 & 16-23, half 1 -> K 8-15 & 24-31
        const int bkb = half ? 8 : 0;
        #pragma unroll
        for (int nf = 0; nf < 8; ++nf) {
            bf16x8 blo = *(const bf16x8*)&Bs[nf * 16 + lr][bkb];
            bf16x8 bhi = *(const bf16x8*)&Bs[nf * 16 + lr][bkb + 16];
            acc[nf] = wmma_bf16(afrag, cat8(blo, bhi), acc[nf]);
        }
        __syncthreads();
    }

    // ---- epilogue: +bias, ReLU, max over this wave's 16 rows ----
    #pragma unroll
    for (int nf = 0; nf < 8; ++nf) {
        const float bcv = bc[n_base + nf * 16 + lr];
        float m = 0.0f;                      // ReLU floor doubles as max identity
        #pragma unroll
        for (int v = 0; v < 8; ++v) m = fmaxf(m, acc[nf][v] + bcv);
        m = fmaxf(m, __shfl_xor(m, 16, 32)); // combine the two 8-row halves
        if (half == 0) red[w][nf * 16 + lr] = m;
    }
    __syncthreads();

    // ---- cross-wave max: sentence 0 = waves 0-3, sentence 1 = waves 4-7 ----
    {
        const int batch = tid >> 7;          // 0 or 1
        const int col   = tid & 127;
        float m = red[batch * 4 + 0][col];
        m = fmaxf(m, red[batch * 4 + 1][col]);
        m = fmaxf(m, red[batch * 4 + 2][col]);
        m = fmaxf(m, red[batch * 4 + 3][col]);
        const int gb = blockIdx.y * 2 + batch;  // global sentence index
        e[(size_t)gb * NC + n_base + col] = m;
    }
}

// =====================================================================
// Stage 1.5: feat = [e1-e2, e1*e2] -> bf16 ; W1 -> bf16
// =====================================================================
__global__ __launch_bounds__(256) void feat_kernel(
    const float* __restrict__ e1, const float* __restrict__ e2,
    bf16_t* __restrict__ feat)
{
    const int i = blockIdx.x * 256 + threadIdx.x;  // over NB * 2C
    const int b = i >> 10;
    const int k = i & 1023;
    const int c = k & (NC - 1);
    const float a = e1[(size_t)b * NC + c];
    const float d = e2[(size_t)b * NC + c];
    feat[i] = (bf16_t)((k < NC) ? (a - d) : (a * d));
}

__global__ __launch_bounds__(256) void cvt_bf16_kernel(
    const float* __restrict__ src, bf16_t* __restrict__ dst, int n)
{
    const int i = blockIdx.x * 256 + threadIdx.x;
    if (i < n) dst[i] = (bf16_t)src[i];
}

__global__ __launch_bounds__(256) void init_out_kernel(
    float* __restrict__ out, const float* __restrict__ b2)
{
    const int i = blockIdx.x * 256 + threadIdx.x;
    if (i < NB) out[i] = b2[0];
}

// =====================================================================
// Stage 2: h = tanh(feat @ W1^T + b1), fused out += h . W2 (atomicAdd)
// TDM double-buffered: wave 0 issues tensor_load_to_lds for tiles k+2
// while all waves compute on tile k; s_wait_tensorcnt(2) = "current
// tile landed, next still in flight".
// grid = (NB/TM, NH/TN), block = 256
// =====================================================================
__global__ __launch_bounds__(256) void mlp_kernel(
    const bf16_t* __restrict__ feat, const bf16_t* __restrict__ W1bf,
    const float* __restrict__ b1, const float* __restrict__ W2,
    float* __restrict__ out)
{
    __shared__ bf16_t As[2][TM][KPAD];
    __shared__ bf16_t Bs[2][TN][KPAD];

    const int tid  = threadIdx.x;
    const int w    = tid >> 5;
    const int l    = tid & 31;
    const int lr   = l & 15;
    const int half = l >> 4;
    const int m_base = blockIdx.x * TM;
    const int n_base = blockIdx.y * TN;
    const int K  = 2 * NC;        // 1024
    const int KS = K / TK;        // 32 k-steps

    const uint32_t ldsA0 = lds_off(&As[0][0][0]);
    const uint32_t ldsA1 = lds_off(&As[1][0][0]);
    const uint32_t ldsB0 = lds_off(&Bs[0][0][0]);
    const uint32_t ldsB1 = lds_off(&Bs[1][0][0]);

    f32x8 acc[8];
    #pragma unroll
    for (int nf = 0; nf < 8; ++nf) acc[nf] = f32x8{};

    // ---- preload k-steps 0 and 1 (wave 0 only: TDM is a wave-level DMA) ----
    if (w == 0) {
        tdm_load_tile32(ldsA0, feat + (size_t)m_base * K + 0 * TK, K, NB);
        tdm_load_tile32(ldsB0, W1bf + (size_t)n_base * K + 0 * TK, K, NH);
        tdm_load_tile32(ldsA1, feat + (size_t)m_base * K + 1 * TK, K, NB);
        tdm_load_tile32(ldsB1, W1bf + (size_t)n_base * K + 1 * TK, K, NH);
    }

    for (int kt = 0; kt < KS; ++kt) {
        const int p = kt & 1;
        if (w == 0) {
            if (kt + 1 < KS) __builtin_amdgcn_s_wait_tensorcnt(2);
            else             __builtin_amdgcn_s_wait_tensorcnt(0);
        }
        __syncthreads();   // releases only after wave0's tensorcnt wait

        const int akb = half ? 8 : 0;
        bf16x8 alo = *(const bf16x8*)&As[p][w * 16 + lr][akb];
        bf16x8 ahi = *(const bf16x8*)&As[p][w * 16 + lr][akb + 16];
        bf16x16 afrag = cat8(alo, ahi);

        const int bkb = half ? 8 : 0;
        #pragma unroll
        for (int nf = 0; nf < 8; ++nf) {
            bf16x8 blo = *(const bf16x8*)&Bs[p][nf * 16 + lr][bkb];
            bf16x8 bhi = *(const bf16x8*)&Bs[p][nf * 16 + lr][bkb + 16];
            acc[nf] = wmma_bf16(afrag, cat8(blo, bhi), acc[nf]);
        }
        __syncthreads();   // all waves done reading buffer p

        if (w == 0 && kt + 2 < KS) {
            const int k2 = (kt + 2) * TK;
            tdm_load_tile32(p ? ldsA1 : ldsA0, feat + (size_t)m_base * K + k2, K, NB);
            tdm_load_tile32(p ? ldsB1 : ldsB0, W1bf + (size_t)n_base * K + k2, K, NH);
        }
    }

    // ---- epilogue: +b1, tanh, dot with W2 over this tile's 128 hidden cols ----
    float pacc[8];
    #pragma unroll
    for (int v = 0; v < 8; ++v) pacc[v] = 0.0f;

    #pragma unroll
    for (int nf = 0; nf < 8; ++nf) {
        const int hc = n_base + nf * 16 + lr;
        const float b1v = b1[hc];
        const float w2v = W2[hc];
        #pragma unroll
        for (int v = 0; v < 8; ++v)
            pacc[v] += tanhf(acc[nf][v] + b1v) * w2v;
    }
    // reduce across the 16 lanes of each half (each VGPR v = one row)
    #pragma unroll
    for (int v = 0; v < 8; ++v) {
        pacc[v] += __shfl_xor(pacc[v], 8, 32);
        pacc[v] += __shfl_xor(pacc[v], 4, 32);
        pacc[v] += __shfl_xor(pacc[v], 2, 32);
        pacc[v] += __shfl_xor(pacc[v], 1, 32);
    }
    if (lr == 0) {
        #pragma unroll
        for (int v = 0; v < 8; ++v) {
            const int row = m_base + w * 16 + v + (half ? 8 : 0);
            atomicAdd(&out[row], pacc[v]);
        }
    }
}

// =====================================================================
// Host-side launch
// inputs: x1, x2, Wc, bc, W1, b1, W2, b2
// ws layout: e1[f32 2048*512] | e2[f32 2048*512] | feat[bf16 2048*1024] | W1bf[bf16 2048*1024]
// =====================================================================
extern "C" void kernel_launch(void* const* d_in, const int* in_sizes, int n_in,
                              void* d_out, int out_size, void* d_ws, size_t ws_size,
                              hipStream_t stream) {
    const float* x1 = (const float*)d_in[0];
    const float* x2 = (const float*)d_in[1];
    const float* Wc = (const float*)d_in[2];
    const float* bc = (const float*)d_in[3];
    const float* W1 = (const float*)d_in[4];
    const float* b1 = (const float*)d_in[5];
    const float* W2 = (const float*)d_in[6];
    const float* b2 = (const float*)d_in[7];
    float* out = (float*)d_out;

    float*  e1   = (float*)d_ws;
    float*  e2   = e1 + (size_t)NB * NC;
    bf16_t* feat = (bf16_t*)(e2 + (size_t)NB * NC);
    bf16_t* W1bf = feat + (size_t)NB * 2 * NC;

    // Stage 1: N-blocks on the fast grid axis so the 4 re-reads of each
    // x-tile stay resident in the 192MB L2.
    dim3 g1(NC / TN, MROWS / TM, 2);       // (4, 1024, 2)
    conv_relu_max_kernel<<<g1, 256, 0, stream>>>(x1, x2, Wc, bc, e1, e2);

    // Stage 1.5: feature build + weight conversion + output init
    feat_kernel<<<(NB * 2 * NC) / 256, 256, 0, stream>>>(e1, e2, feat);
    cvt_bf16_kernel<<<(NH * 2 * NC + 255) / 256, 256, 0, stream>>>(W1, W1bf, NH * 2 * NC);
    init_out_kernel<<<(NB + 255) / 256, 256, 0, stream>>>(out, b2);

    // Stage 2: fused tanh-MLP + final dot, TDM-fed
    mlp_kernel<<<dim3(NB / TM, NH / TN), 256, 0, stream>>>(feat, W1bf, b1, W2, out);
}
